// IPAGNN_35270271434819
// MI455X (gfx1250) — matrix operand order repaired and checked by reading, not compiled, not run on hardware
//
#include <hip/hip_runtime.h>
#include <hip/hip_bf16.h>
#include <math.h>

// ---------------------------------------------------------------------------
// Problem constants (match reference)
// ---------------------------------------------------------------------------
#define BB_    16          // batch
#define NN_    256         // nodes
#define TT_    4           // tokens
#define HH_    128         // hidden
#define LL_    2           // lstm layers
#define GG_    512         // 4*H gate width
#define MM_    (BB_*NN_)   // 4096 flattened nodes
#define MH_    (MM_*HH_)   // 524288
#define VOCAB_ 1000
#define OUTV_  1000
#define STEPS_ 15

typedef __attribute__((ext_vector_type(16))) __bf16 v16bf;
typedef __attribute__((ext_vector_type(8)))  __bf16 v8bf;
typedef __attribute__((ext_vector_type(8)))  float  v8f;

// ---------------------------------------------------------------------------
// Weight repack: W (128 x 512 row-major, f32) -> bf16 in contiguous-k (k = 16h + p) B-fragment
// order for V_WMMA_F32_16X16X32_BF16.  B is 32x16 (KxN) per fragment:
//   lane = n%16 + 16*((k%32)>>4), element p = k%16, frag = (k/32, n/16).
// Each lane then loads its 16 bf16 (32B) contiguously.
// ---------------------------------------------------------------------------
__global__ void k_pack_w(const float* __restrict__ W, __bf16* __restrict__ out) {
    int idx = blockIdx.x * blockDim.x + threadIdx.x;        // over 128*512
    if (idx >= HH_ * GG_) return;
    int k = idx >> 9, n = idx & 511;
    int k32 = k >> 5, kk = k & 31;
    int nt  = n >> 4, nnl = n & 15;
    int lane = nnl + ((kk >> 4) << 4);
    int p    = kk & 15;
    int dst  = ((k32 * (GG_ >> 4) + nt) << 9) + (lane << 4) + p;
    out[dst] = (__bf16)W[idx];
}

// embT[t*M*H + m*H + h] = bf16(embed[data[m*T+t] * H + h])
__global__ void k_embed(const int* __restrict__ data, const float* __restrict__ embed,
                        __bf16* __restrict__ embT) {
    int idx = blockIdx.x * blockDim.x + threadIdx.x;        // over T*M*H
    int h = idx & (HH_ - 1);
    int m = (idx >> 7) & (MM_ - 1);
    int t = idx >> 19;                                      // M*H = 2^19
    int tok = data[m * TT_ + t];
    embT[idx] = (__bf16)embed[tok * HH_ + h];
}

__global__ void k_init_ip(float* __restrict__ ip) {
    int idx = blockIdx.x * blockDim.x + threadIdx.x;        // over M
    if (idx < MM_) ip[idx] = ((idx & (NN_ - 1)) == 0) ? 1.0f : 0.0f;
}

// ---------------------------------------------------------------------------
// WMMA GEMM:  C[M x 512] = (Cinit | bias | 0) + A0[M x 128] @ B0 (+ A1 @ B1)
// A: bf16 row-major (K=128).  B: pre-packed bf16 fragments.  C: f32.
// Block = 256 threads (8 waves), tile 64 rows x 128 cols; wave = 16 x 64.
// grid = (G/128, M/64)
// ---------------------------------------------------------------------------
__global__ __launch_bounds__(256)
void k_gemm(const __bf16* __restrict__ A0, const __bf16* __restrict__ B0,
            const __bf16* __restrict__ A1, const __bf16* __restrict__ B1,
            const float* __restrict__ Cinit, const float* __restrict__ bias,
            float* __restrict__ C) {
    constexpr int K = HH_;
    const int wave = threadIdx.x >> 5;
    const int lane = threadIdx.x & 31;
    const int rowBase = blockIdx.y * 64 + (wave >> 1) * 16;
    const int colBase = blockIdx.x * 128 + (wave & 1) * 64;
    const int kh = lane >> 4;          // k-half of this lane (A frag / C rows)
    const int ml = lane & 15;          // row within A frag / col within C tile
    const int nt0 = colBase >> 4;
    const int ntiles = GG_ >> 4;

    v8f acc[4];
#pragma unroll
    for (int t = 0; t < 4; ++t) {
#pragma unroll
        for (int j = 0; j < 8; ++j) {
            int r = rowBase + kh * 8 + j;
            int c = colBase + t * 16 + ml;
            float v = 0.0f;
            if (Cinit) v = Cinit[(size_t)r * GG_ + c];
            if (bias)  v += bias[c];
            acc[t][j] = v;
        }
    }

#pragma unroll
    for (int k32 = 0; k32 < K / 32; ++k32) {
        // A fragment, pair 0: lane holds row ml, K = {kh*16..+15}
        const __bf16* ap = A0 + (size_t)(rowBase + ml) * K + k32 * 32 + kh * 16;
        if (k32 + 1 < K / 32) __builtin_prefetch(ap + 32, 0, 3);
        v8bf alo = *(const v8bf*)ap;
        v8bf ahi = *(const v8bf*)(ap + 8);
        v16bf a;
#pragma unroll
        for (int i = 0; i < 8; ++i) { a[i] = alo[i]; a[i + 8] = ahi[i]; }
#pragma unroll
        for (int t = 0; t < 4; ++t) {
            const v16bf b = *(const v16bf*)(B0 + (size_t)(k32 * ntiles + nt0 + t) * 512
                                               + lane * 16);
            acc[t] = __builtin_amdgcn_wmma_f32_16x16x32_bf16(
                         false, a, false, b, (short)0, acc[t], false, false);
        }
        if (A1) {
            const __bf16* ap1 = A1 + (size_t)(rowBase + ml) * K + k32 * 32 + kh * 16;
            v8bf alo1 = *(const v8bf*)ap1;
            v8bf ahi1 = *(const v8bf*)(ap1 + 8);
            v16bf a1;
#pragma unroll
            for (int i = 0; i < 8; ++i) { a1[i] = alo1[i]; a1[i + 8] = ahi1[i]; }
#pragma unroll
            for (int t = 0; t < 4; ++t) {
                const v16bf b1 = *(const v16bf*)(B1 + (size_t)(k32 * ntiles + nt0 + t) * 512
                                                    + lane * 16);
                acc[t] = __builtin_amdgcn_wmma_f32_16x16x32_bf16(
                             false, a1, false, b1, (short)0, acc[t], false, false);
            }
        }
    }

#pragma unroll
    for (int t = 0; t < 4; ++t)
#pragma unroll
        for (int j = 0; j < 8; ++j) {
            int r = rowBase + kh * 8 + j;
            int c = colBase + t * 16 + ml;
            C[(size_t)r * GG_ + c] = acc[t][j];
        }
}

// ---------------------------------------------------------------------------
// LSTM gates:  z = [i|f|g|o] (M x 512);  c,h updated in place; hbf = bf16(h)
// ---------------------------------------------------------------------------
__device__ __forceinline__ float sigm(float x) { return 1.0f / (1.0f + expf(-x)); }

__global__ void k_gates(const float* __restrict__ z, float* __restrict__ c,
                        float* __restrict__ h, __bf16* __restrict__ hbf) {
    int idx = blockIdx.x * blockDim.x + threadIdx.x;        // over M*H
    int m = idx >> 7, hh = idx & (HH_ - 1);
    const float* zm = z + (size_t)m * GG_;
    float zi = zm[hh], zf = zm[HH_ + hh], zg = zm[2 * HH_ + hh], zo = zm[3 * HH_ + hh];
    float cv = c[idx];
    float c2 = sigm(zf) * cv + sigm(zi) * tanhf(zg);
    float h2 = sigm(zo) * tanhf(c2);
    c[idx] = c2; h[idx] = h2; hbf[idx] = (__bf16)h2;
}

// Copy cur -> work, and refresh bf16 h mirrors for the GEMM A operands.
__global__ void k_step_begin(const float* __restrict__ cc0, const float* __restrict__ ch0,
                             const float* __restrict__ cc1, const float* __restrict__ ch1,
                             float* __restrict__ wc0, float* __restrict__ wh0,
                             float* __restrict__ wc1, float* __restrict__ wh1,
                             __bf16* __restrict__ h0bf, __bf16* __restrict__ h1bf) {
    int idx = blockIdx.x * blockDim.x + threadIdx.x;        // over M*H
    float a = cc0[idx], b = ch0[idx], c = cc1[idx], d = ch1[idx];
    wc0[idx] = a; wh0[idx] = b; wc1[idx] = c; wh1[idx] = d;
    h0bf[idx] = (__bf16)b; h1bf[idx] = (__bf16)d;
}

// Restore exit-node state:  work[:, b, exit[b]] = cur[:, b, exit[b]]
__global__ void k_restore(const int* __restrict__ exitI,
                          const float* __restrict__ cc0, const float* __restrict__ ch0,
                          const float* __restrict__ cc1, const float* __restrict__ ch1,
                          float* __restrict__ wc0, float* __restrict__ wh0,
                          float* __restrict__ wc1, float* __restrict__ wh1) {
    int b = blockIdx.x, h = threadIdx.x;
    int i = (b * NN_ + exitI[b]) * HH_ + h;
    wc0[i] = cc0[i]; wh0[i] = ch0[i]; wc1[i] = cc1[i]; wh1[i] = ch1[i];
}

// Branch softmax:  feats = [c0|h0|c1|h1] (512) per node; wt/wf = p * ip
__global__ void k_branch(const float* __restrict__ wc0, const float* __restrict__ wh0,
                         const float* __restrict__ wc1, const float* __restrict__ wh1,
                         const float* __restrict__ Wb, const float* __restrict__ bbv,
                         const float* __restrict__ ip,
                         float* __restrict__ wt, float* __restrict__ wf) {
    __shared__ float s0[128], s1[128];
    int m = blockIdx.x, t = threadIdx.x;
    size_t o = (size_t)m * HH_ + t;
    float f0 = wc0[o], f1 = wh0[o], f2 = wc1[o], f3 = wh1[o];
    float p0 = f0 * Wb[2 * t]           + f1 * Wb[2 * (t + 128)]
             + f2 * Wb[2 * (t + 256)]   + f3 * Wb[2 * (t + 384)];
    float p1 = f0 * Wb[2 * t + 1]       + f1 * Wb[2 * (t + 128) + 1]
             + f2 * Wb[2 * (t + 256) + 1] + f3 * Wb[2 * (t + 384) + 1];
    s0[t] = p0; s1[t] = p1; __syncthreads();
    for (int s = 64; s > 0; s >>= 1) {
        if (t < s) { s0[t] += s0[t + s]; s1[t] += s1[t + s]; }
        __syncthreads();
    }
    if (t == 0) {
        float l0 = s0[0] + bbv[0], l1 = s1[0] + bbv[1];
        float mx = fmaxf(l0, l1);
        float e0 = expf(l0 - mx), e1 = expf(l1 - mx);
        float inv = 1.0f / (e0 + e1);
        float ipm = ip[m];
        wt[m] = e0 * inv * ipm;
        wf[m] = e1 * inv * ipm;
    }
}

// Weighted scatter-add of node states and IP mass (float global atomics)
__global__ void k_scatter(const int* __restrict__ tbn, const int* __restrict__ fbn,
                          const float* __restrict__ wt, const float* __restrict__ wf,
                          const float* __restrict__ w0, const float* __restrict__ w1,
                          const float* __restrict__ w2, const float* __restrict__ w3,
                          float* __restrict__ a0, float* __restrict__ a1,
                          float* __restrict__ a2, float* __restrict__ a3,
                          float* __restrict__ ipnew) {
    int m = blockIdx.x, h = threadIdx.x;
    float wtm = wt[m], wfm = wf[m];
    if (wtm == 0.0f && wfm == 0.0f) return;   // dead node: zero contribution
    int b = m >> 8;                           // N = 256
    int td = (b * NN_ + tbn[m]) * HH_ + h;
    int fd = (b * NN_ + fbn[m]) * HH_ + h;
    size_t s = (size_t)m * HH_ + h;
    float v0 = w0[s], v1 = w1[s], v2 = w2[s], v3 = w3[s];
    atomicAdd(&a0[td], wtm * v0); atomicAdd(&a0[fd], wfm * v0);
    atomicAdd(&a1[td], wtm * v1); atomicAdd(&a1[fd], wfm * v1);
    atomicAdd(&a2[td], wtm * v2); atomicAdd(&a2[fd], wfm * v2);
    atomicAdd(&a3[td], wtm * v3); atomicAdd(&a3[fd], wfm * v3);
    if (h == 0) {
        atomicAdd(&ipnew[b * NN_ + tbn[m]], wtm);
        atomicAdd(&ipnew[b * NN_ + fbn[m]], wfm);
    }
}

// Live-masked state update:  cur = agg / (ipnew + 1e-7); ip = ipnew (if live)
__global__ void k_finalize(float* __restrict__ cc0, float* __restrict__ ch0,
                           float* __restrict__ cc1, float* __restrict__ ch1,
                           const float* __restrict__ a0, const float* __restrict__ a1,
                           const float* __restrict__ a2, const float* __restrict__ a3,
                           float* __restrict__ ip, const float* __restrict__ ipnew,
                           const int* __restrict__ steps, int step) {
    int idx = blockIdx.x * blockDim.x + threadIdx.x;        // over M*H
    int m = idx >> 7, b = m >> 8;
    if (step < steps[b]) {
        float inv = 1.0f / (ipnew[m] + 1e-7f);
        cc0[idx] = a0[idx] * inv; ch0[idx] = a1[idx] * inv;
        cc1[idx] = a2[idx] * inv; ch1[idx] = a3[idx] * inv;
        if ((idx & (HH_ - 1)) == 0) ip[m] = ipnew[m];
    }
}

// Final logits: feats(b) = exit-node [c0|h0|c1|h1];  out = feats @ Wo + bo
__global__ void k_logits(const float* __restrict__ cc0, const float* __restrict__ ch0,
                         const float* __restrict__ cc1, const float* __restrict__ ch1,
                         const int* __restrict__ exitI, const float* __restrict__ Wo,
                         const float* __restrict__ bo, float* __restrict__ out) {
    __shared__ float feats[2 * LL_ * HH_];                  // 512
    int b = blockIdx.y;
    int m = b * NN_ + exitI[b];
    for (int j = threadIdx.x; j < 2 * LL_ * HH_; j += blockDim.x) {
        int seg = j >> 7, d = j & (HH_ - 1);
        size_t s = (size_t)m * HH_ + d;
        feats[j] = (seg == 0) ? cc0[s] : (seg == 1) ? ch0[s] : (seg == 2) ? cc1[s] : ch1[s];
    }
    __syncthreads();
    int o = blockIdx.x * blockDim.x + threadIdx.x;
    if (o < OUTV_) {
        float acc = bo[o];
#pragma unroll 4
        for (int j = 0; j < 2 * LL_ * HH_; ++j) acc += feats[j] * Wo[(size_t)j * OUTV_ + o];
        out[(size_t)b * OUTV_ + o] = acc;
    }
}

// ---------------------------------------------------------------------------
// Host orchestration
// ---------------------------------------------------------------------------
extern "C" void kernel_launch(void* const* d_in, const int* in_sizes, int n_in,
                              void* d_out, int out_size, void* d_ws, size_t ws_size,
                              hipStream_t stream) {
    const int*   data  = (const int*)  d_in[0];
    const int*   tbn   = (const int*)  d_in[1];
    const int*   fbn   = (const int*)  d_in[2];
    const int*   exitI = (const int*)  d_in[3];
    const int*   steps = (const int*)  d_in[4];
    const float* embed = (const float*)d_in[5];
    const float* Wi    = (const float*)d_in[6];   // (L,128,512)
    const float* Wh    = (const float*)d_in[7];   // (L,128,512)
    const float* bl    = (const float*)d_in[8];   // (L,512)
    const float* Wb    = (const float*)d_in[9];   // (512,2)
    const float* bbv   = (const float*)d_in[10];  // (2,)
    const float* Wo    = (const float*)d_in[11];  // (512,1000)
    const float* bo    = (const float*)d_in[12];  // (1000,)
    float* out = (float*)d_out;

    char* base = (char*)d_ws; size_t off = 0;
    auto carve = [&](size_t bytes) -> void* {
        off = (off + 255) & ~(size_t)255;
        void* p = base + off; off += bytes; return p;
    };
    __bf16* embT = (__bf16*)carve((size_t)TT_ * MM_ * HH_ * 2);
    __bf16* WiP  = (__bf16*)carve((size_t)LL_ * HH_ * GG_ * 2);
    __bf16* WhP  = (__bf16*)carve((size_t)LL_ * HH_ * GG_ * 2);
    float*  Zx0  = (float*) carve((size_t)TT_ * MM_ * GG_ * 4);
    float*  z    = (float*) carve((size_t)MM_ * GG_ * 4);
    float*  cc0  = (float*) carve((size_t)MH_ * 4);
    float*  ch0  = (float*) carve((size_t)MH_ * 4);
    float*  cc1  = (float*) carve((size_t)MH_ * 4);
    float*  ch1  = (float*) carve((size_t)MH_ * 4);
    float*  wc0  = (float*) carve((size_t)MH_ * 4);
    float*  wh0  = (float*) carve((size_t)MH_ * 4);
    float*  wc1  = (float*) carve((size_t)MH_ * 4);
    float*  wh1  = (float*) carve((size_t)MH_ * 4);
    float*  a0   = (float*) carve((size_t)MH_ * 4);
    float*  a1   = (float*) carve((size_t)MH_ * 4);
    float*  a2   = (float*) carve((size_t)MH_ * 4);
    float*  a3   = (float*) carve((size_t)MH_ * 4);
    __bf16* h0bf = (__bf16*)carve((size_t)MH_ * 2);
    __bf16* h1bf = (__bf16*)carve((size_t)MH_ * 2);
    float*  ip    = (float*)carve((size_t)MM_ * 4);
    float*  ipnew = (float*)carve((size_t)MM_ * 4);
    float*  wt    = (float*)carve((size_t)MM_ * 4);
    float*  wf    = (float*)carve((size_t)MM_ * 4);

    const dim3 gemmGrid(GG_ / 128, MM_ / 64);
    const int HG = HH_ * GG_;

    // --- one-time prep (weights repack, embedding gather, Zx0 precompute) ---
    k_pack_w<<<(HG + 255) / 256, 256, 0, stream>>>(Wi,      WiP);
    k_pack_w<<<(HG + 255) / 256, 256, 0, stream>>>(Wi + HG, WiP + HG);
    k_pack_w<<<(HG + 255) / 256, 256, 0, stream>>>(Wh,      WhP);
    k_pack_w<<<(HG + 255) / 256, 256, 0, stream>>>(Wh + HG, WhP + HG);
    k_embed <<<(TT_ * MH_) / 256, 256, 0, stream>>>(data, embed, embT);

    hipMemsetAsync(cc0, 0, (size_t)MH_ * 4, stream);
    hipMemsetAsync(ch0, 0, (size_t)MH_ * 4, stream);
    hipMemsetAsync(cc1, 0, (size_t)MH_ * 4, stream);
    hipMemsetAsync(ch1, 0, (size_t)MH_ * 4, stream);
    k_init_ip<<<MM_ / 256, 256, 0, stream>>>(ip);

    for (int t = 0; t < TT_; ++t)   // Zx0[t] = emb_t @ Wi0 + b0  (reused 15x)
        k_gemm<<<gemmGrid, 256, 0, stream>>>(embT + (size_t)t * MH_, WiP,
                                             nullptr, nullptr, nullptr, bl,
                                             Zx0 + (size_t)t * MM_ * GG_);

    // --- 15 outer interpreter steps ---
    for (int step = 0; step < STEPS_; ++step) {
        k_step_begin<<<MH_ / 256, 256, 0, stream>>>(cc0, ch0, cc1, ch1,
                                                    wc0, wh0, wc1, wh1, h0bf, h1bf);
        for (int t = 0; t < TT_; ++t) {
            // layer 0:  z = Zx0[t] + h0 @ Wh0
            k_gemm<<<gemmGrid, 256, 0, stream>>>(h0bf, WhP, nullptr, nullptr,
                                                 Zx0 + (size_t)t * MM_ * GG_, nullptr, z);
            k_gates<<<MH_ / 256, 256, 0, stream>>>(z, wc0, wh0, h0bf);
            // layer 1:  z = b1 + h0_new @ Wi1 + h1 @ Wh1   (fused dual-pair GEMM)
            k_gemm<<<gemmGrid, 256, 0, stream>>>(h0bf, WiP + HG, h1bf, WhP + HG,
                                                 nullptr, bl + GG_, z);
            k_gates<<<MH_ / 256, 256, 0, stream>>>(z, wc1, wh1, h1bf);
        }
        k_restore<<<BB_, HH_, 0, stream>>>(exitI, cc0, ch0, cc1, ch1,
                                           wc0, wh0, wc1, wh1);
        k_branch<<<MM_, HH_, 0, stream>>>(wc0, wh0, wc1, wh1, Wb, bbv, ip, wt, wf);
        hipMemsetAsync(a0, 0, (size_t)MH_ * 4, stream);
        hipMemsetAsync(a1, 0, (size_t)MH_ * 4, stream);
        hipMemsetAsync(a2, 0, (size_t)MH_ * 4, stream);
        hipMemsetAsync(a3, 0, (size_t)MH_ * 4, stream);
        hipMemsetAsync(ipnew, 0, (size_t)MM_ * 4, stream);
        k_scatter<<<MM_, HH_, 0, stream>>>(tbn, fbn, wt, wf, wc0, wh0, wc1, wh1,
                                           a0, a1, a2, a3, ipnew);
        k_finalize<<<MH_ / 256, 256, 0, stream>>>(cc0, ch0, cc1, ch1,
                                                  a0, a1, a2, a3, ip, ipnew,
                                                  steps, step);
    }

    // --- final logits (B,1,1000) ---
    k_logits<<<dim3((OUTV_ + 255) / 256, BB_), 256, 0, stream>>>(
        cc0, ch0, cc1, ch1, exitI, Wo, bo, out);
    (void)in_sizes; (void)n_in; (void)out_size; (void)ws_size;
}
